// Get_Correlation_31284541784121
// MI455X (gfx1250) — matrix operand order repaired and checked by reading, not compiled, not run on hardware
//
#include <hip/hip_runtime.h>

typedef __attribute__((ext_vector_type(16))) __bf16 v16bf;
typedef __attribute__((ext_vector_type(4)))  __bf16 v4bf;
typedef __attribute__((ext_vector_type(8)))  float  v8f;

#define BB   2
#define CC   256
#define TT   16
#define HWD  784
#define NTHW (TT * HWD)   // 12544
#define RCC  16
#define SDT  32           // sd (key) tile
#define NSDT 25           // ceil(784/32)

// Store K-index k (0..31) at pos(k) so that the 16 A-operand elements of
// lane-half h (K = a_kidx(e,h)) are the contiguous run [16h, 16h+16).
__device__ __forceinline__ int kpos(int k) {
  return (k & 7) | (((k >> 4) & 1) << 3) | (((k >> 3) & 1) << 4);
}

__device__ __forceinline__ v8f wmma_bf16(v16bf a, v16bf b, v8f c) {
  return __builtin_amdgcn_wmma_f32_16x16x32_bf16(false, a, false, b,
                                                 (short)0, c, false, false);
}

// ---------------------------------------------------------------------------
// x2 = W_down2 (256x256) @ x (256 x 12544) per batch -> bf16, x-layout.
// 8 waves; wave w owns M-tiles {2w,2w+1} x 4 N-tiles (N block = 64).
// Wl is stored A-fragment-ready (pos-permuted K), Xl B-fragment-ready.
// ---------------------------------------------------------------------------
__global__ void __launch_bounds__(256)
k_gemm_x2(const float* __restrict__ x, const float* __restrict__ Wd2,
          __bf16* __restrict__ x2bf) {
  __shared__ __attribute__((aligned(64))) __bf16 Wl[256][48]; // [m][pos-k]
  __shared__ __attribute__((aligned(64))) __bf16 Xl[64][32];  // [n][pos-k]
  const int b   = blockIdx.y;
  const int n0  = blockIdx.x * 64;
  const int tid = threadIdx.x;
  const int w = tid >> 5, lane = tid & 31;
  const int col = lane & 15, half = lane >> 4;

  const v8f vzero = {0.f, 0.f, 0.f, 0.f, 0.f, 0.f, 0.f, 0.f};
  v8f acc[2][4];
#pragma unroll
  for (int i = 0; i < 2; ++i)
#pragma unroll
    for (int j = 0; j < 4; ++j) acc[i][j] = vzero;

  for (int kc = 0; kc < 8; ++kc) {
    __syncthreads();
    for (int idx = tid; idx < 32 * 256; idx += 256) {  // coalesced over k
      int m = idx >> 5, k = idx & 31;
      Wl[m][kpos(k)] = (__bf16)Wd2[m * 256 + kc * 32 + k];
    }
    for (int idx = tid; idx < 32 * 64; idx += 256) {   // coalesced over n
      int k = idx >> 6, n = idx & 63;
      Xl[n][kpos(k)] = (__bf16)x[(b * CC + kc * 32 + k) * NTHW + n0 + n];
    }
    __syncthreads();

    v16bf bv[4];
#pragma unroll
    for (int nt = 0; nt < 4; ++nt)
      bv[nt] = *(const v16bf*)&Xl[nt * 16 + col][half * 16];
#pragma unroll
    for (int mi = 0; mi < 2; ++mi) {
      v16bf a = *(const v16bf*)&Wl[(w * 2 + mi) * 16 + col][half * 16];
#pragma unroll
      for (int nt = 0; nt < 4; ++nt)
        acc[mi][nt] = wmma_bf16(a, bv[nt], acc[mi][nt]);
    }
  }
#pragma unroll
  for (int mi = 0; mi < 2; ++mi)
#pragma unroll
    for (int nt = 0; nt < 4; ++nt)
#pragma unroll
      for (int r = 0; r < 8; ++r) {
        int o = (w * 2 + mi) * 16 + r + 8 * half;
        int n = n0 + nt * 16 + col;
        x2bf[(b * CC + o) * NTHW + n] = (__bf16)acc[mi][nt][r];
      }
}

// ---------------------------------------------------------------------------
// Fused dual-branch attention-like correlation:
//   S[hw,sd] = sum_c x[c,hw]*x2f[c,sd]; g = wbr*(sigmoid(S)-0.5)
//   O[hw,c] += sum_sd g[hw,sd]*x2f[c,sd]   (streamed over 25 sd-tiles of 32)
// Q A-fragments resident in VGPRs (staged once through LDS, coalesced).
// wave w: mt = w>>1 (hw subtile), GEMM1 nt = w&1, GEMM2 c-half jj = w&1.
// ---------------------------------------------------------------------------
__global__ void __launch_bounds__(256)
k_corr(const float* __restrict__ x, const __bf16* __restrict__ x2bf,
       const float* __restrict__ w2, float* __restrict__ feat) {
  __shared__ __attribute__((aligned(64))) __bf16 Ks[64][272]; // [sd][pos-c] / Q stage [hw][pos-c]
  __shared__ __attribute__((aligned(64))) __bf16 Kt[256][48]; // [c][pos-sd]
  __shared__ __attribute__((aligned(64))) __bf16 Ss[64][32];  // [hw][pos-sd]
  const int b = blockIdx.z, t = blockIdx.y;
  const int hw0 = blockIdx.x * 64;
  const int tid = threadIdx.x;
  const int w = tid >> 5, lane = tid & 31;
  const int col = lane & 15, half = lane >> 4;
  const int mt = w >> 1;
  const int jj = w & 1;

  // ---- stage Q tile [hw][pos-permuted c] (coalesced over hw), build frags
  for (int idx = tid; idx < 64 * 256; idx += 256) {
    int c = idx >> 6, m = idx & 63;
    int hw = hw0 + m;
    float v = (hw < HWD) ? x[((b * CC + c) * TT + t) * HWD + hw] : 0.f;
    Ks[m][(c & ~31) + kpos(c & 31)] = (__bf16)v;
  }
  __syncthreads();
  v16bf aq[8];
#pragma unroll
  for (int kc = 0; kc < 8; ++kc)
    aq[kc] = *(const v16bf*)&Ks[mt * 16 + col][kc * 32 + half * 16];

  const v8f vzero = {0.f, 0.f, 0.f, 0.f, 0.f, 0.f, 0.f, 0.f};
  v8f acc[8];
#pragma unroll
  for (int i = 0; i < 8; ++i) acc[i] = vzero;

  for (int br = 0; br < 2; ++br) {
    const float wbr = w2[br];
    const int tf = (br == 0) ? ((t + 1 < TT) ? t + 1 : TT - 1)
                             : ((t > 0) ? t - 1 : 0);
    for (int sdt = 0; sdt < NSDT; ++sdt) {
      const int sd0 = sdt * SDT;
      __syncthreads();  // prev iter / Q-frag build done with Ks, Ss
      // dual-orientation tile load: b64 global reads, b64 stores to Kt,
      // b16 scatter to Ks. 784 % 4 == 0 so quads never straddle validity.
      for (int it = tid; it < (SDT * 256) / 4; it += 256) {
        int c = it >> 3;
        int s4 = (it & 7) * 4;
        int sd = sd0 + s4;
        v4bf q;
        if (sd < HWD) {
          q = *(const v4bf*)&x2bf[((b * CC + c) * TT + tf) * HWD + sd];
        } else {
          q = (v4bf)(__bf16)0.f;
        }
        *(v4bf*)&Kt[c][kpos(s4)] = q;
#pragma unroll
        for (int qq = 0; qq < 4; ++qq) Ks[s4 + qq][(c & ~31) + kpos(c & 31)] = q[qq];
      }
      __syncthreads();

      // GEMM1: wave computes S tile (mt, nt = jj), K = C = 256
      {
        const int nt = jj;
        v8f s = vzero;
#pragma unroll
        for (int kc = 0; kc < 8; ++kc) {
          v16bf bk = *(const v16bf*)&Ks[nt * 16 + col][kc * 32 + half * 16];
          s = wmma_bf16(aq[kc], bk, s);
        }
        const int pc = kpos(nt * 16 + col);
#pragma unroll
        for (int r = 0; r < 8; ++r) {
          float g = wbr * (1.f / (1.f + __expf(-s[r])) - 0.5f);
          Ss[mt * 16 + r + 8 * half][pc] = (__bf16)g;
        }
      }
      __syncthreads();

      // GEMM2: O[mt, c-tiles of jj half] += g(S) * V, K = SDT = 32
      {
        v16bf aS = *(const v16bf*)&Ss[mt * 16 + col][half * 16];
#pragma unroll
        for (int ct = 0; ct < 8; ++ct) {
          v16bf bV = *(const v16bf*)&Kt[(jj * 8 + ct) * 16 + col][half * 16];
          acc[ct] = wmma_bf16(aS, bV, acc[ct]);
        }
      }
    }
  }

#pragma unroll
  for (int ct = 0; ct < 8; ++ct) {
    int cidx = (jj * 8 + ct) * 16 + col;
#pragma unroll
    for (int r = 0; r < 8; ++r) {
      int m = mt * 16 + r + 8 * half;
      int hw = hw0 + m;
      if (hw < HWD)
        feat[((b * CC + cidx) * TT + t) * HWD + hw] = acc[ct][r];
    }
  }
}

// ---------------------------------------------------------------------------
// Gating branch (tiny, VALU): xr = W_down @ x
// ---------------------------------------------------------------------------
__global__ void __launch_bounds__(256)
k_down(const float* __restrict__ x, const float* __restrict__ Wd,
       float* __restrict__ xr) {
  int idx = blockIdx.x * 256 + threadIdx.x;
  if (idx >= BB * RCC * NTHW) return;
  int n = idx % NTHW;
  int o = (idx / NTHW) % RCC;
  int b = idx / (NTHW * RCC);
  float s = 0.f;
  for (int c = 0; c < CC; ++c)
    s += Wd[o * CC + c] * x[(b * CC + c) * NTHW + n];
  xr[idx] = s;
}

// Three dilated depthwise 9x3x3 convs, weighted sum.
__global__ void __launch_bounds__(256)
k_dwconv(const float* __restrict__ xr, const float* __restrict__ K1,
         const float* __restrict__ K2, const float* __restrict__ K3,
         const float* __restrict__ wts, float* __restrict__ agg) {
  int idx = blockIdx.x * 256 + threadIdx.x;
  if (idx >= BB * RCC * NTHW) return;
  int hw = idx % HWD;
  int t  = (idx / HWD) % TT;
  int rc = (idx / (HWD * TT)) % RCC;
  int b  = idx / (HWD * TT * RCC);
  int h = hw / 28, wq = hw % 28;
  const float* Kj3[3] = {K1, K2, K3};
  const float* base = xr + (b * RCC + rc) * NTHW;
  float total = 0.f;
  for (int j = 0; j < 3; ++j) {
    const float* Kj = Kj3[j] + rc * 81;
    int dil = j + 1;
    float s = 0.f;
    for (int kt = 0; kt < 9; ++kt) {
      int tt2 = t + kt - 4;
      if (tt2 < 0 || tt2 >= TT) continue;
      for (int kh = 0; kh < 3; ++kh) {
        int hh = h + (kh - 1) * dil;
        if (hh < 0 || hh >= 28) continue;
        for (int kw = 0; kw < 3; ++kw) {
          int ww = wq + (kw - 1) * dil;
          if (ww < 0 || ww >= 28) continue;
          s += Kj[kt * 9 + kh * 3 + kw] * base[tt2 * HWD + hh * 28 + ww];
        }
      }
    }
    total += wts[j] * s;
  }
  agg[idx] = total;
}

// out = features1 * (sigmoid(W_back @ agg) - 0.5)
__global__ void __launch_bounds__(256)
k_final(const float* __restrict__ feat, const float* __restrict__ agg,
        const float* __restrict__ Wb, float* __restrict__ out) {
  int idx = blockIdx.x * 256 + threadIdx.x;
  if (idx >= BB * CC * NTHW) return;
  int n = idx % NTHW;
  int o = (idx / NTHW) % CC;
  int b = idx / (NTHW * CC);
  float s = 0.f;
#pragma unroll
  for (int rc = 0; rc < RCC; ++rc)
    s += Wb[o * RCC + rc] * agg[(b * RCC + rc) * NTHW + n];
  float gate = 1.f / (1.f + __expf(-s)) - 0.5f;
  out[idx] = feat[idx] * gate;
}

// ---------------------------------------------------------------------------
extern "C" void kernel_launch(void* const* d_in, const int* in_sizes, int n_in,
                              void* d_out, int out_size, void* d_ws,
                              size_t ws_size, hipStream_t stream) {
  const float* x   = (const float*)d_in[0];
  const float* Wd  = (const float*)d_in[1];
  const float* Wd2 = (const float*)d_in[2];
  const float* K1  = (const float*)d_in[3];
  const float* K2  = (const float*)d_in[4];
  const float* K3  = (const float*)d_in[5];
  const float* Wb  = (const float*)d_in[6];
  const float* wts = (const float*)d_in[7];
  const float* w2  = (const float*)d_in[8];
  float* out = (float*)d_out;

  // workspace carve-up (256B aligned):
  //   x2 bf16  : 12,845,056 B
  //   features : 25,690,112 B
  //   xr       :  1,605,632 B
  //   agg      :  1,605,632 B     (total ~41.7 MB)
  char* ws = (char*)d_ws;
  __bf16* x2bf = (__bf16*)ws;
  float*  feat = (float*)(ws + 12845056);
  float*  xr   = (float*)(ws + 12845056 + 25690112);
  float*  agg  = (float*)(ws + 12845056 + 25690112 + 1605632);

  k_gemm_x2<<<dim3(196, BB), 256, 0, stream>>>(x, Wd2, x2bf);
  k_down<<<dim3((BB * RCC * NTHW + 255) / 256), 256, 0, stream>>>(x, Wd, xr);
  k_dwconv<<<dim3((BB * RCC * NTHW + 255) / 256), 256, 0, stream>>>(
      xr, K1, K2, K3, wts, agg);
  k_corr<<<dim3(13, TT, BB), 256, 0, stream>>>(x, x2bf, w2, feat);
  k_final<<<dim3((BB * CC * NTHW) / 256), 256, 0, stream>>>(feat, agg, Wb, out);
}
